// GNNModel_72404558676146
// MI455X (gfx1250) — compile-verified, matrix operand order checked by reading, not run on hardware
//
#include <hip/hip_runtime.h>
#include <hip/hip_bf16.h>

#define NN    20000
#define EE    160000
#define HH    8

typedef __attribute__((ext_vector_type(16))) __bf16 v16bf;
typedef __attribute__((ext_vector_type(8)))  float  v8f;

union FragBF {
    v16bf v;
    float4 q[2];
};

// ---------- helpers ----------
__device__ __forceinline__ unsigned short f2bf(float f) {
    unsigned u = __float_as_uint(f);
    unsigned r = u + 0x7fffu + ((u >> 16) & 1u);   // round-to-nearest-even
    return (unsigned short)(r >> 16);
}
__device__ __forceinline__ unsigned encF(float f) {   // order-preserving uint encoding
    unsigned u = __float_as_uint(f);
    return (u & 0x80000000u) ? ~u : (u | 0x80000000u);
}
__device__ __forceinline__ float decF(unsigned u) {
    return (u & 0x80000000u) ? __uint_as_float(u & 0x7fffffffu) : __uint_as_float(~u);
}
__device__ __forceinline__ float gelu_exact(float x) {
    return 0.5f * x * (1.0f + erff(x * 0.70710678118654752f));
}

// ---------- conversion / init kernels ----------
__global__ void k_f32_to_bf16(const float* __restrict__ in, unsigned short* __restrict__ out, int n) {
    int i = blockIdx.x * blockDim.x + threadIdx.x;
    if (i < n) out[i] = f2bf(in[i]);
}

// Wt[nn*K + k] = W[k*Nn + nn]  (transpose + convert), count = Nn*K
__global__ void k_transpose_bf16(const float* __restrict__ W, unsigned short* __restrict__ Wt,
                                 int K, int Nn) {
    int i = blockIdx.x * blockDim.x + threadIdx.x;
    if (i >= Nn * K) return;
    int nn = i / K;
    int k  = i - nn * K;
    Wt[i] = f2bf(W[(size_t)k * Nn + nn]);
}

// zero n4 float4's
__global__ void k_zero_f32_v4(float4* __restrict__ p, int n4) {
    int i = blockIdx.x * blockDim.x + threadIdx.x;
    if (i < n4) p[i] = make_float4(0.f, 0.f, 0.f, 0.f);
}

__global__ void k_init_softmax(unsigned* __restrict__ emaxU, float* __restrict__ denom, int n) {
    int i = blockIdx.x * blockDim.x + threadIdx.x;
    if (i < n) { emaxU[i] = encF(-__builtin_inff()); denom[i] = 0.0f; }
}

// ---------- WMMA GEMM: C[M,Nn] = A[M,K](bf16) * Wt[Nn,K](bf16)^T ----------
// Register-blocked: each wave computes a 16Mx64N strip (4 accumulators),
// reusing one A fragment across 4 B fragments per K step (2.5 loads / wmma).
// Block = 8 waves (2 wave-rows x 4 wave-cols) -> 32M x 256N per block.
// grid = (Nn/256, M/32), block = 256.
__global__ void k_gemm_bf16_wmma(const unsigned short* __restrict__ A,
                                 const unsigned short* __restrict__ Bt,
                                 float* __restrict__ C,
                                 int K, int Nn) {
    const int lane = threadIdx.x & 31;
    const int wave = threadIdx.x >> 5;
    const int wrow = wave >> 2;          // 0..1
    const int wcol = wave & 3;           // 0..3
    const int tileM = blockIdx.y * 32 + wrow * 16;
    const int tileN = blockIdx.x * 256 + wcol * 64;

    const int halfsel = lane >> 4;       // 0: lanes 0-15, 1: lanes 16-31
    const int klo     = halfsel * 8;     // K sub-offset per ISA 16-bit A/B layout
    const int rowA    = tileM + (lane & 15);

    const unsigned short* ap = A + (size_t)rowA * K + klo;
    const unsigned short* bp[4];
#pragma unroll
    for (int j = 0; j < 4; ++j)
        bp[j] = Bt + (size_t)(tileN + 16 * j + (lane & 15)) * K + klo;

    v8f c0 = {}, c1 = {}, c2 = {}, c3 = {};
    for (int kb = 0; kb < K; kb += 32) {
        // prefetch the K tile two steps ahead in the A stream (lowers to global_prefetch_b8)
        __builtin_prefetch((const void*)(ap + kb + 64), 0, 1);

        FragBF a;
        a.q[0] = *(const float4*)(ap + kb);        // K = kb+klo    .. +7
        a.q[1] = *(const float4*)(ap + kb + 16);   // K = kb+16+klo .. +7

        FragBF b0, b1, b2, b3;
        b0.q[0] = *(const float4*)(bp[0] + kb);  b0.q[1] = *(const float4*)(bp[0] + kb + 16);
        b1.q[0] = *(const float4*)(bp[1] + kb);  b1.q[1] = *(const float4*)(bp[1] + kb + 16);
        b2.q[0] = *(const float4*)(bp[2] + kb);  b2.q[1] = *(const float4*)(bp[2] + kb + 16);
        b3.q[0] = *(const float4*)(bp[3] + kb);  b3.q[1] = *(const float4*)(bp[3] + kb + 16);

        c0 = __builtin_amdgcn_wmma_f32_16x16x32_bf16(false, a.v, false, b0.v, (short)0, c0, false, false);
        c1 = __builtin_amdgcn_wmma_f32_16x16x32_bf16(false, a.v, false, b1.v, (short)0, c1, false, false);
        c2 = __builtin_amdgcn_wmma_f32_16x16x32_bf16(false, a.v, false, b2.v, (short)0, c2, false, false);
        c3 = __builtin_amdgcn_wmma_f32_16x16x32_bf16(false, a.v, false, b3.v, (short)0, c3, false, false);
    }

    // 32-bit C/D layout: VGPR r -> row tileM + r + 8*halfsel, col = tileN + (lane&15)
    const int outRow = tileM + halfsel * 8;
    float* cbase = C + (size_t)outRow * Nn + tileN + (lane & 15);
    v8f acc[4] = { c0, c1, c2, c3 };
#pragma unroll
    for (int j = 0; j < 4; ++j) {
        float* cp = cbase + 16 * j;
#pragma unroll
        for (int r = 0; r < 8; ++r)
            cp[(size_t)r * Nn] = acc[j][r];
    }
}

// ---------- attention logits: el/er[n,h] = sum_f feat[n,h,f]*a[h,f] ----------
// one wave per (n,h); grid = N*H/8, block = 256
__global__ void k_attn_logits(const float* __restrict__ feat,
                              const float* __restrict__ al,
                              const float* __restrict__ ar,
                              float* __restrict__ el, float* __restrict__ er, int F) {
    const int lane = threadIdx.x & 31;
    const int wave = threadIdx.x >> 5;
    const int task = blockIdx.x * 8 + wave;
    if (task >= NN * HH) return;
    const int n = task >> 3, h = task & 7;
    const float* fp  = feat + (size_t)n * (HH * F) + h * F;
    const float* alp = al + h * F;
    const float* arp = ar + h * F;
    float sl = 0.0f, sr = 0.0f;
    for (int f = lane; f < F; f += 32) {
        float v = fp[f];
        sl += v * alp[f];
        sr += v * arp[f];
    }
#pragma unroll
    for (int m = 16; m; m >>= 1) {
        sl += __shfl_xor(sl, m, 32);
        sr += __shfl_xor(sr, m, 32);
    }
    if (lane == 0) { el[task] = sl; er[task] = sr; }
}

// ---------- edge logits + segment max ----------
__global__ void k_edge_logits(const int* __restrict__ src, const int* __restrict__ dst,
                              const float* __restrict__ el, const float* __restrict__ er,
                              float* __restrict__ elog, unsigned* __restrict__ emaxU) {
    int i = blockIdx.x * blockDim.x + threadIdx.x;
    if (i >= EE * HH) return;
    int e = i >> 3, h = i & 7;
    int s = src[e], d = dst[e];
    float x = el[s * HH + h] + er[d * HH + h];
    x = (x > 0.0f) ? x : 0.2f * x;                 // leaky relu, slope 0.2
    elog[i] = x;
    atomicMax(&emaxU[d * HH + h], encF(x));
}

// ---------- exp + segment sum (ee written in place over elog) ----------
__global__ void k_edge_exp(const int* __restrict__ dst,
                           float* __restrict__ elog,
                           const unsigned* __restrict__ emaxU,
                           float* __restrict__ denom) {
    int i = blockIdx.x * blockDim.x + threadIdx.x;
    if (i >= EE * HH) return;
    int e = i >> 3, h = i & 7;
    int d = dst[e];
    float m  = decF(emaxU[d * HH + h]);
    float ee = __expf(elog[i] - m);
    elog[i] = ee;
    atomicAdd(&denom[d * HH + h], ee);
}

// ---------- scatter-sum: out[dst,h,:] += alpha * feat[src,h,:] ----------
// one wave per (e,h); float4 gathers, scalar f32 atomics; grid = E*H/8
__global__ void k_aggregate(const int* __restrict__ src, const int* __restrict__ dst,
                            const float* __restrict__ ee, const float* __restrict__ denom,
                            const float* __restrict__ feat, float* __restrict__ out, int F) {
    const int lane = threadIdx.x & 31;
    const int wave = threadIdx.x >> 5;
    const int task = blockIdx.x * 8 + wave;
    if (task >= EE * HH) return;
    const int e = task >> 3, h = task & 7;
    const int s = src[e], d = dst[e];
    const float alpha = ee[task] / denom[d * HH + h];
    const float* fp = feat + (size_t)s * (HH * F) + h * F;
    float* op       = out  + (size_t)d * (HH * F) + h * F;
    for (int f = lane * 4; f < F; f += 128) {      // F is a multiple of 128
        float4 v = *(const float4*)(fp + f);
        atomicAdd(&op[f + 0], alpha * v.x);
        atomicAdd(&op[f + 1], alpha * v.y);
        atomicAdd(&op[f + 2], alpha * v.z);
        atomicAdd(&op[f + 3], alpha * v.w);
    }
}

// ---------- mid-layer finalize: h[n,f] = mean_h gelu(agg[n,h,f] + b[h,f]) ----------
__global__ void k_finalize_mid(const float* __restrict__ agg, const float* __restrict__ b,
                               float* __restrict__ hbuf, int F) {
    int i = blockIdx.x * blockDim.x + threadIdx.x;
    if (i >= NN * F) return;
    int n = i / F, f = i - n * F;
    float s = 0.0f;
#pragma unroll
    for (int h = 0; h < HH; ++h) {
        float x = agg[(size_t)n * (HH * F) + h * F + f] + b[h * F + f];
        s += gelu_exact(x);
    }
    hbuf[i] = s * (1.0f / HH);
}

// ---------- final layer: out[n,h,f] = agg[n,h,f] + b[h,f] ----------
__global__ void k_finalize_out(const float* __restrict__ agg, const float* __restrict__ b,
                               float* __restrict__ out, int HF) {
    int i = blockIdx.x * blockDim.x + threadIdx.x;
    if (i >= NN * HF) return;
    int j = i % HF;
    out[i] = agg[i] + b[j];
}

// ================= host side =================
extern "C" void kernel_launch(void* const* d_in, const int* in_sizes, int n_in,
                              void* d_out, int out_size, void* d_ws, size_t ws_size,
                              hipStream_t stream) {
    (void)in_sizes; (void)n_in; (void)out_size; (void)ws_size;

    const float* node = (const float*)d_in[0];
    const int*   src  = (const int*)d_in[1];
    const int*   dst  = (const int*)d_in[2];
    const float* W[3]  = { (const float*)d_in[3],  (const float*)d_in[7],  (const float*)d_in[11] };
    const float* al[3] = { (const float*)d_in[4],  (const float*)d_in[8],  (const float*)d_in[12] };
    const float* ar[3] = { (const float*)d_in[5],  (const float*)d_in[9],  (const float*)d_in[13] };
    const float* bb[3] = { (const float*)d_in[6],  (const float*)d_in[10], (const float*)d_in[14] };
    const int Kdim[3] = { 512, 256, 256 };
    const int Fdim[3] = { 256, 256, 512 };

    // ---- workspace partition (256-byte aligned slabs) ----
    char* ws = (char*)d_ws;
    size_t off = 0;
    auto take = [&](size_t bytes) -> char* {
        char* p = ws + off;
        off += (bytes + 255) & ~(size_t)255;
        return p;
    };
    unsigned short* xbf  = (unsigned short*)take((size_t)NN * 512 * 2);          // layer input, bf16
    unsigned short* wtbf = (unsigned short*)take((size_t)4096 * 512 * 2);        // Wt, bf16
    float* feat   = (float*)take((size_t)NN * 4096 * 4);                         // GEMM output (L3 size)
    float* hbuf   = (float*)take((size_t)NN * 256 * 4);                          // merged hidden
    float* el     = (float*)take((size_t)NN * HH * 4);
    float* er     = (float*)take((size_t)NN * HH * 4);
    unsigned* emaxU = (unsigned*)take((size_t)NN * HH * 4);
    float* denom  = (float*)take((size_t)NN * HH * 4);
    float* elog   = (float*)take((size_t)EE * HH * 4);
    // L1/L2 aggregation buffer aliases the unused upper half of the L3-sized feat slab
    float* aggMid = feat + (size_t)NN * 2048;

    const int BT = 256;
    for (int l = 0; l < 3; ++l) {
        const int K  = Kdim[l];
        const int F  = Fdim[l];
        const int HF = HH * F;
        const float* xin = (l == 0) ? node : hbuf;
        float* agg = (l == 2) ? (float*)d_out : aggMid;

        // 1) input -> bf16
        {
            int n = NN * K;
            k_f32_to_bf16<<<(n + BT - 1) / BT, BT, 0, stream>>>(xin, xbf, n);
        }
        // 2) W -> bf16 transposed [HF, K]
        {
            int n = HF * K;
            k_transpose_bf16<<<(n + BT - 1) / BT, BT, 0, stream>>>(W[l], wtbf, K, HF);
        }
        // 3) GEMM via WMMA bf16 (register-blocked 16x64 per wave): feat[N, HF]
        {
            dim3 grid(HF / 256, NN / 32);
            k_gemm_bf16_wmma<<<grid, BT, 0, stream>>>(xbf, wtbf, feat, K, HF);
        }
        // 4) attention logits
        k_attn_logits<<<NN * HH / 8, BT, 0, stream>>>(feat, al[l], ar[l], el, er, F);
        // 5) softmax state init
        {
            int n = NN * HH;
            k_init_softmax<<<(n + BT - 1) / BT, BT, 0, stream>>>(emaxU, denom, n);
        }
        // 6) edge logits + segment max
        {
            int n = EE * HH;
            k_edge_logits<<<(n + BT - 1) / BT, BT, 0, stream>>>(src, dst, el, er, elog, emaxU);
            // 7) exp + segment sum
            k_edge_exp<<<(n + BT - 1) / BT, BT, 0, stream>>>(dst, elog, emaxU, denom);
        }
        // 8) zero aggregation target (float4 stores)
        {
            int n4 = NN * HF / 4;
            k_zero_f32_v4<<<(n4 + BT - 1) / BT, BT, 0, stream>>>((float4*)agg, n4);
        }
        // 9) scatter-sum messages
        k_aggregate<<<EE * HH / 8, BT, 0, stream>>>(src, dst, elog, denom, feat, agg, F);
        // 10) finalize
        if (l < 2) {
            int n = NN * F;
            k_finalize_mid<<<(n + BT - 1) / BT, BT, 0, stream>>>(agg, bb[l], hbuf, F);
        } else {
            int n = NN * HF;
            k_finalize_out<<<(n + BT - 1) / BT, BT, 0, stream>>>(agg, bb[l], (float*)d_out, HF);
        }
    }
}
